// SwinTransformerBlock3D_65206193488029
// MI455X (gfx1250) — matrix operand order repaired and checked by reading, not compiled, not run on hardware
//
#include <hip/hip_runtime.h>
#include <hip/hip_fp16.h>

typedef _Float16 h16;
typedef __attribute__((ext_vector_type(16))) _Float16 v16h;
typedef __attribute__((ext_vector_type(8)))  float    v8f;

union FragA { v16h v; uint4 q[2]; };

#define EPI_F16            0
#define EPI_F32_RES        1
#define EPI_F32_REMAP_RES  2

__device__ __forceinline__ float waveRedSum(float v) {
  for (int m = 16; m >= 1; m >>= 1) v += __shfl_xor(v, m, 32);
  return v;
}
__device__ __forceinline__ float halfRedMax(float v) {
  for (int m = 8; m >= 1; m >>= 1) v = fmaxf(v, __shfl_xor(v, m, 32));
  return v;
}
__device__ __forceinline__ float halfRedSum(float v) {
  for (int m = 8; m >= 1; m >>= 1) v += __shfl_xor(v, m, 32);
  return v;
}

// ---------------------------------------------------------------------------
// gfx1250 async global->LDS copy (ASYNCcnt path), CDNA5 ISA
// ---------------------------------------------------------------------------
__device__ __forceinline__ void async_b128(const void* gptr, unsigned lofs) {
  asm volatile("global_load_async_to_lds_b128 %0, %1, off"
               :: "v"(lofs), "v"(gptr) : "memory");
}
__device__ __forceinline__ unsigned lds_off(const void* lptr) {
  return (unsigned)(size_t)lptr;  // generic LDS ptr: low 32 bits = LDS offset
}
__device__ __forceinline__ void wait_async0() {
  asm volatile("s_wait_asynccnt 0" ::: "memory");
}

// ---------------------------------------------------------------------------
// Weight conversion f32 -> f16
// ---------------------------------------------------------------------------
__global__ void f32_to_f16_kernel(const float* __restrict__ in, h16* __restrict__ out, int n) {
  int i = blockIdx.x * 256 + threadIdx.x;
  if (i < n) out[i] = (h16)in[i];
}

// ---------------------------------------------------------------------------
// LayerNorm1 + roll(-1,-3,-3) + window partition -> f16 rows (window order)
// ---------------------------------------------------------------------------
__global__ __launch_bounds__(256)
void ln1_window_kernel(const float* __restrict__ x,
                       const float* __restrict__ w, const float* __restrict__ b,
                       h16* __restrict__ XW) {
  int wrow = blockIdx.x * 8 + (threadIdx.x >> 5);
  int lane = threadIdx.x & 31;
  int win = wrow / 98, n = wrow % 98;
  int bb = win >> 8;
  int r  = win & 255;
  int dz = r >> 6, hy = (r >> 3) & 7, wx = r & 7;
  int ld = n / 49, lh = (n % 49) / 7, lw = n % 7;
  int d  = (dz * 2 + ld + 1) & 7;
  int hh = (hy * 7 + lh + 3) % 56;
  int ww = (wx * 7 + lw + 3) % 56;
  const float* px = x + (size_t)(((bb * 8 + d) * 56 + hh) * 56 + ww) * 384;

  float vals[12]; float s = 0.f;
  for (int i = 0; i < 12; ++i) { vals[i] = px[lane + i * 32]; s += vals[i]; }
  s = waveRedSum(s);
  float mean = s * (1.f / 384.f);
  float vs = 0.f;
  for (int i = 0; i < 12; ++i) { float t = vals[i] - mean; vs += t * t; }
  vs = waveRedSum(vs);
  float inv = rsqrtf(vs * (1.f / 384.f) + 1e-5f);

  h16* po = XW + (size_t)wrow * 384;
  for (int i = 0; i < 12; ++i) {
    int c = lane + i * 32;
    po[c] = (h16)((vals[i] - mean) * inv * w[c] + b[c]);
  }
}

// ---------------------------------------------------------------------------
// LayerNorm2 (token order) -> f16
// ---------------------------------------------------------------------------
__global__ __launch_bounds__(256)
void ln2_kernel(const float* __restrict__ x1,
                const float* __restrict__ w, const float* __restrict__ b,
                h16* __restrict__ Y) {
  int t = blockIdx.x * 8 + (threadIdx.x >> 5);
  int lane = threadIdx.x & 31;
  const float* px = x1 + (size_t)t * 384;
  float vals[12]; float s = 0.f;
  for (int i = 0; i < 12; ++i) { vals[i] = px[lane + i * 32]; s += vals[i]; }
  s = waveRedSum(s);
  float mean = s * (1.f / 384.f);
  float vs = 0.f;
  for (int i = 0; i < 12; ++i) { float t2 = vals[i] - mean; vs += t2 * t2; }
  vs = waveRedSum(vs);
  float inv = rsqrtf(vs * (1.f / 384.f) + 1e-5f);
  h16* po = Y + (size_t)t * 384;
  for (int i = 0; i < 12; ++i) {
    int c = lane + i * 32;
    po[c] = (h16)((vals[i] - mean) * inv * w[c] + b[c]);
  }
}

// ---------------------------------------------------------------------------
// WMMA GEMM: D[M,N] = A[M,K](f16) * W[N,K]^T(f16) + bias, optional GELU.
// 8 waves (4M x 2N), wave tile 32Mx64N (8 WMMA/K-step), block 128Mx128N,
// K-step 32, double-buffered LDS via global_load_async_to_lds_b128.
// K/32 is even for all call sites, so the K-loop is unrolled x2 with
// statically-assigned ping/pong buffers (no phi/select), and prefetch is
// unconditional (caller guarantees a 256B tail pad after every operand).
// ---------------------------------------------------------------------------
__global__ __launch_bounds__(256)
void gemm_wmma_kernel(const h16* __restrict__ A, const h16* __restrict__ Bw,
                      const float* __restrict__ bias,
                      int M, int N, int K, int mode, int gelu,
                      h16* __restrict__ outH, float* __restrict__ outF,
                      const float* __restrict__ res) {
  __shared__ __align__(16) h16 As[2][128 * 32];
  __shared__ __align__(16) h16 Bs[2][128 * 32];
  int tid  = threadIdx.x;
  int lane = tid & 31, wv = tid >> 5;
  int wm = wv >> 1, wn = wv & 1;        // 4 x 2 wave grid
  int hi = lane >> 4, ln16 = lane & 15;
  size_t m0 = (size_t)blockIdx.y * 128;
  size_t n0 = (size_t)blockIdx.x * 128;

  v8f acc[2][4];
  for (int i = 0; i < 2; ++i)
    for (int j = 0; j < 4; ++j)
      for (int e = 0; e < 8; ++e) acc[i][j][e] = 0.f;

  int rS = tid >> 2, cS = (tid & 3) * 8;      // 16B chunk per thread, x2 rows

  const h16* gA0 = A  + (m0 + rS)      * (size_t)K + cS;
  const h16* gA1 = A  + (m0 + rS + 64) * (size_t)K + cS;
  const h16* gB0 = Bw + (n0 + rS)      * (size_t)K + cS;
  const h16* gB1 = Bw + (n0 + rS + 64) * (size_t)K + cS;
  const unsigned lA0_0 = lds_off(&As[0][rS * 32 + cS]);
  const unsigned lA1_0 = lds_off(&As[0][(rS + 64) * 32 + cS]);
  const unsigned lB0_0 = lds_off(&Bs[0][rS * 32 + cS]);
  const unsigned lB1_0 = lds_off(&Bs[0][(rS + 64) * 32 + cS]);
  const unsigned lA0_1 = lds_off(&As[1][rS * 32 + cS]);
  const unsigned lA1_1 = lds_off(&As[1][(rS + 64) * 32 + cS]);
  const unsigned lB0_1 = lds_off(&Bs[1][rS * 32 + cS]);
  const unsigned lB1_1 = lds_off(&Bs[1][(rS + 64) * 32 + cS]);

  auto advance = [&]() { gA0 += 32; gA1 += 32; gB0 += 32; gB1 += 32; };
  auto stage = [&](unsigned a0, unsigned a1, unsigned b0, unsigned b1) {
    async_b128(gA0, a0); async_b128(gA1, a1);
    async_b128(gB0, b0); async_b128(gB1, b1);
  };
  auto compute = [&](const h16* ab, const h16* bb) {
    FragA af[2];
    for (int i = 0; i < 2; ++i) {
      const uint4* p = (const uint4*)(ab + (wm * 32 + i * 16 + ln16) * 32);
      af[i].q[0] = p[hi]; af[i].q[1] = p[2 + hi];
    }
    for (int j = 0; j < 4; ++j) {
      FragA bf;
      const uint4* p = (const uint4*)(bb + (wn * 64 + j * 16 + ln16) * 32);
      bf.q[0] = p[hi]; bf.q[1] = p[2 + hi];
      for (int i = 0; i < 2; ++i)
        acc[i][j] = __builtin_amdgcn_wmma_f32_16x16x32_f16(
            false, af[i].v, false, bf.v, (short)0, acc[i][j], false, false);
    }
  };

  // prologue: stage k-block 0 into buffer 0
  stage(lA0_0, lA1_0, lB0_0, lB1_0);
  wait_async0();
  __syncthreads();

  int nsteps = K >> 5;  // always even at all call sites
  for (int s = 0; s < nsteps; s += 2) {
    advance();
    stage(lA0_1, lA1_1, lB0_1, lB1_1);   // prefetch next (last one is dead)
    compute(As[0], Bs[0]);
    wait_async0();
    __syncthreads();

    advance();
    stage(lA0_0, lA1_0, lB0_0, lB1_0);
    compute(As[1], Bs[1]);
    wait_async0();
    __syncthreads();
  }

  for (int j = 0; j < 4; ++j) {
    size_t n = n0 + wn * 64 + j * 16 + ln16;
    float bn = bias[n];
    for (int i = 0; i < 2; ++i) {
      for (int r = 0; r < 8; ++r) {
        size_t m = m0 + wm * 32 + i * 16 + hi * 8 + r;
        float v = acc[i][j][r] + bn;
        if (gelu) v = 0.5f * v * (1.f + erff(v * 0.70710678118654752f));
        if (mode == EPI_F16) {
          outH[m * N + n] = (h16)v;
        } else if (mode == EPI_F32_RES) {
          outF[m * N + n] = v + res[m * N + n];
        } else { // EPI_F32_REMAP_RES: window row -> token row (reverse+roll)
          int wrow = (int)m;
          int win = wrow / 98, nn = wrow % 98;
          int bb2 = win >> 8, rr = win & 255;
          int dz = rr >> 6, hy = (rr >> 3) & 7, wx = rr & 7;
          int ld = nn / 49, lh = (nn % 49) / 7, lw = nn % 7;
          int d  = (dz * 2 + ld + 1) & 7;
          int hh = (hy * 7 + lh + 3) % 56;
          int ww = (wx * 7 + lw + 3) % 56;
          size_t tr = (size_t)(((bb2 * 8 + d) * 56 + hh) * 56 + ww);
          outF[tr * N + n] = v + res[tr * N + n];
        }
      }
    }
  }
}

// ---------------------------------------------------------------------------
// Windowed attention: one block per (window, head). N=98 padded to 112.
// Q/K staged via async global->LDS copies (scale folded into epilogue),
// S = QKt via WMMA + rel-pos bias + shift mask (on the fly), softmax f32,
// O = P V via WMMA, f16 out in window-row order.
// ---------------------------------------------------------------------------
__global__ __launch_bounds__(256)
void attn_kernel(const h16* __restrict__ QKV, const float* __restrict__ rpb,
                 h16* __restrict__ OUT) {
  __shared__ __align__(16) h16 Qs[112 * 32];
  __shared__ __align__(16) h16 Ks[112 * 32];
  __shared__ __align__(16) h16 VTs[32 * 128];
  __shared__ __align__(16) h16 Ps[112 * 128];

  int win = blockIdx.x, head = blockIdx.y;
  int tid = threadIdx.x, lane = tid & 31, wv = tid >> 5;
  int hi = lane >> 4, ln16 = lane & 15;
  const float scale = 0.17677669529663687f; // 1/sqrt(32)

  size_t rowBase = (size_t)win * 98;
  const h16* q0 = QKV + rowBase * 1152 + head * 32;
  const h16* k0 = q0 + 384;
  const h16* v0 = q0 + 768;

  for (int i = 98 * 32 + tid; i < 112 * 32; i += 256) { Qs[i] = (h16)0.f; Ks[i] = (h16)0.f; }
  for (int ch = tid; ch < 98 * 4; ch += 256) {
    int n = ch >> 2, c = (ch & 3) * 8;
    async_b128(q0 + (size_t)n * 1152 + c, lds_off(Qs + n * 32 + c));
    async_b128(k0 + (size_t)n * 1152 + c, lds_off(Ks + n * 32 + c));
  }
  for (int i = tid; i < 32 * 128; i += 256) {
    int d = i >> 7, n = i & 127;
    float vv = (n < 98) ? (float)v0[(size_t)n * 1152 + d] : 0.f;
    VTs[d * 128 + n] = (h16)vv;
  }
  for (int i = tid; i < 112 * 128; i += 256) Ps[i] = (h16)0.f;
  wait_async0();
  __syncthreads();

  int rr = win & 255;
  int dz = rr >> 6, hy = (rr >> 3) & 7, wx = rr & 7;

  if (wv < 7) {
    v8f accS[7];
    for (int j = 0; j < 7; ++j) for (int e = 0; e < 8; ++e) accS[j][e] = 0.f;
    FragA af;
    {
      const uint4* pq = (const uint4*)(Qs + (wv * 16 + ln16) * 32);
      af.q[0] = pq[hi]; af.q[1] = pq[2 + hi];
    }
    for (int j = 0; j < 7; ++j) {
      FragA bf;
      const uint4* pk = (const uint4*)(Ks + (j * 16 + ln16) * 32);
      bf.q[0] = pk[hi]; bf.q[1] = pk[2 + hi];
      accS[j] = __builtin_amdgcn_wmma_f32_16x16x32_f16(
          false, af.v, false, bf.v, (short)0, accS[j], false, false);
    }

    float sreg[7][8];
    for (int j = 0; j < 7; ++j) {
      int c = j * 16 + ln16;
      bool cok = c < 98;
      int d2 = 0, h2 = 0, w2 = 0, lab2 = 0;
      if (cok) {
        d2 = c / 49; h2 = (c % 49) / 7; w2 = c % 7;
        int gd = dz * 2 + d2, gh = hy * 7 + h2, gw = wx * 7 + w2;
        int rd = gd < 6 ? 0 : (gd < 7 ? 1 : 2);
        int rh = gh < 49 ? 0 : (gh < 53 ? 1 : 2);
        int rw = gw < 49 ? 0 : (gw < 53 ? 1 : 2);
        lab2 = rd * 9 + rh * 3 + rw;
      }
      for (int r8 = 0; r8 < 8; ++r8) {
        int row = wv * 16 + hi * 8 + r8;
        float s = -1e30f;
        if (cok && row < 98) {
          int d1 = row / 49, h1 = (row % 49) / 7, w1 = row % 7;
          int gd = dz * 2 + d1, gh = hy * 7 + h1, gw = wx * 7 + w1;
          int rd = gd < 6 ? 0 : (gd < 7 ? 1 : 2);
          int rh = gh < 49 ? 0 : (gh < 53 ? 1 : 2);
          int rw = gw < 49 ? 0 : (gw < 53 ? 1 : 2);
          int lab1 = rd * 9 + rh * 3 + rw;
          int idx = (d1 - d2 + 1) * 169 + (h1 - h2 + 6) * 13 + (w1 - w2 + 6);
          s = accS[j][r8] * scale + rpb[idx * 12 + head] + ((lab1 != lab2) ? -100.f : 0.f);
        }
        sreg[j][r8] = s;
      }
    }
    for (int r8 = 0; r8 < 8; ++r8) {
      float m = -1e30f;
      for (int j = 0; j < 7; ++j) m = fmaxf(m, sreg[j][r8]);
      m = halfRedMax(m);
      float e[7]; float sum = 0.f;
      for (int j = 0; j < 7; ++j) { e[j] = __expf(sreg[j][r8] - m); sum += e[j]; }
      sum = halfRedSum(sum);
      float inv = 1.f / sum;
      int row = wv * 16 + hi * 8 + r8;
      for (int j = 0; j < 7; ++j)
        Ps[row * 128 + j * 16 + ln16] = (h16)(e[j] * inv);
    }
  }
  __syncthreads();

  for (int t = wv; t < 14; t += 8) {
    int mt = t >> 1, nt = t & 1;
    v8f acc;
    for (int e = 0; e < 8; ++e) acc[e] = 0.f;
    for (int ks = 0; ks < 4; ++ks) {
      FragA af, bf;
      const uint4* pp = (const uint4*)(Ps + (mt * 16 + ln16) * 128 + ks * 32);
      af.q[0] = pp[hi]; af.q[1] = pp[2 + hi];
      const uint4* pv = (const uint4*)(VTs + (nt * 16 + ln16) * 128 + ks * 32);
      bf.q[0] = pv[hi]; bf.q[1] = pv[2 + hi];
      acc = __builtin_amdgcn_wmma_f32_16x16x32_f16(
          false, af.v, false, bf.v, (short)0, acc, false, false);
    }
    for (int r8 = 0; r8 < 8; ++r8) {
      int row = mt * 16 + hi * 8 + r8;
      if (row < 98)
        OUT[(rowBase + row) * 384 + head * 32 + nt * 16 + ln16] = (h16)acc[r8];
    }
  }
}

// ---------------------------------------------------------------------------
extern "C" void kernel_launch(void* const* d_in, const int* in_sizes, int n_in,
                              void* d_out, int out_size, void* d_ws, size_t ws_size,
                              hipStream_t stream) {
  const float* x      = (const float*)d_in[0];
  const float* n1w    = (const float*)d_in[1];
  const float* n1b    = (const float*)d_in[2];
  const float* qkv_w  = (const float*)d_in[3];
  const float* qkv_b  = (const float*)d_in[4];
  const float* rpb    = (const float*)d_in[5];
  const float* proj_w = (const float*)d_in[6];
  const float* proj_b = (const float*)d_in[7];
  const float* n2w    = (const float*)d_in[8];
  const float* n2b    = (const float*)d_in[9];
  const float* fc1_w  = (const float*)d_in[10];
  const float* fc1_b  = (const float*)d_in[11];
  const float* fc2_w  = (const float*)d_in[12];
  const float* fc2_b  = (const float*)d_in[13];
  float* out = (float*)d_out;
  (void)in_sizes; (void)n_in; (void)out_size; (void)ws_size;

  const size_t M = 50176;  // 2*8*56*56 tokens = 512 windows * 98
  char* ws = (char*)d_ws;
  size_t off = 0;
  auto take = [&](size_t bytes) -> char* {
    char* p = ws + off;
    off = (off + bytes + 255) & ~(size_t)255;
    return p;
  };
  h16*   XW  = (h16*)take(M * 384 * 2);
  h16*   QKV = (h16*)take(M * 1152 * 2);
  h16*   ATT = (h16*)take(M * 384 * 2);
  float* X1  = (float*)take(M * 384 * 4);
  h16*   Y   = (h16*)take(M * 384 * 2);
  h16*   Hb  = (h16*)take(M * 1536 * 2);
  h16*   Wq  = (h16*)take((size_t)1152 * 384 * 2);
  h16*   Wp  = (h16*)take((size_t)384 * 384 * 2);
  h16*   W1  = (h16*)take((size_t)1536 * 384 * 2);
  h16*   W2  = (h16*)take((size_t)384 * 1536 * 2);
  (void)take(256);  // tail pad: makes the GEMM's unconditional prefetch safe

  auto conv = [&](const float* s, h16* d, int n) {
    f32_to_f16_kernel<<<(n + 255) / 256, 256, 0, stream>>>(s, d, n);
  };
  conv(qkv_w, Wq, 1152 * 384);
  conv(proj_w, Wp, 384 * 384);
  conv(fc1_w, W1, 1536 * 384);
  conv(fc2_w, W2, 384 * 1536);

  ln1_window_kernel<<<M / 8, 256, 0, stream>>>(x, n1w, n1b, XW);

  gemm_wmma_kernel<<<dim3(1152 / 128, M / 128), 256, 0, stream>>>(
      XW, Wq, qkv_b, (int)M, 1152, 384, EPI_F16, 0, QKV, nullptr, nullptr);

  attn_kernel<<<dim3(512, 12), 256, 0, stream>>>(QKV, rpb, ATT);

  gemm_wmma_kernel<<<dim3(384 / 128, M / 128), 256, 0, stream>>>(
      ATT, Wp, proj_b, (int)M, 384, 384, EPI_F32_REMAP_RES, 0, nullptr, X1, x);

  ln2_kernel<<<M / 8, 256, 0, stream>>>(X1, n2w, n2b, Y);

  gemm_wmma_kernel<<<dim3(1536 / 128, M / 128), 256, 0, stream>>>(
      Y, W1, fc1_b, (int)M, 1536, 384, EPI_F16, 1, Hb, nullptr, nullptr);

  gemm_wmma_kernel<<<dim3(384 / 128, M / 128), 256, 0, stream>>>(
      Hb, W2, fc2_b, (int)M, 384, 1536, EPI_F32_RES, 0, nullptr, out, X1);
}
